// Def_DownSampling_58420145160697
// MI455X (gfx1250) — matrix operand order repaired and checked by reading, not compiled, not run on hardware
//
#include <hip/hip_runtime.h>
#include <hip/hip_bf16.h>
#include <stdint.h>

typedef _Float16 v16h __attribute__((ext_vector_type(16)));
typedef float    v8f  __attribute__((ext_vector_type(8)));

// ---------------- problem geometry ----------------
#define Y_ELEMS  2097152   // (2,32,32,32,32)
#define D_ELEMS  2097152   // (2,32,32,32,32)
#define G_ELEMS  196608    // (2,32,32,32,3)

// ---------------- LDS layout for the fused conv+pool kernel ----------------
// input tile: [z:4][y:6][x:34][ch slots: 34 f16 (32 data + 2 pad)] -> 17-uint lane stride (conflict-free)
#define IN_ZE 4
#define IN_YE 6
#define IN_XE 34
#define PT_H  34
#define PT_U  17
#define IN_H  (IN_ZE*IN_YE*IN_XE*PT_H)     // 27744 f16 = 55488 B
// weights as pre-swizzled A fragments: [tap:27][mtile:2][lane:32] x 18 f16 (16 data + 2 pad) -> 9-uint stride
#define WFRAG_H 18
#define WFRAG_U 9
#define NFRAG   (27*2*32)                  // 1728
#define W_H     (NFRAG*WFRAG_H)            // 31104 f16 = 62208 B
#define WOFF_H  IN_H
#define WOFF_U  (IN_H/2)
#define LDS_BYTES ((IN_H + W_H)*2)         // 117696 B (32KB pool buffer aliases input region)

// =====================================================================
// Kernel 1: conv3d 3x3x3 (32->32) + bias + fused 2x2x2 avgpool, via
// implicit GEMM with v_wmma_f32_16x16x32_f16.  2048 blocks x 256 thr.
// block = (batch, pooled z, pooled-y pair, x half)
// =====================================================================
__global__ __launch_bounds__(256) void conv_pool_wmma(
    const float* __restrict__ x, const float* __restrict__ cw,
    const float* __restrict__ cb, float* __restrict__ Dout)
{
  extern __shared__ char smem[];
  _Float16* lh = (_Float16*)smem;
  uint32_t* lu = (uint32_t*)smem;

  const int tid = threadIdx.x;
  const int bi  = blockIdx.x;
  const int bxh = bi & 1;            // x half: conv x base = bxh*32
  const int by  = (bi >> 1) & 15;    // pooled y pair: conv y base = by*4
  const int bz  = (bi >> 5) & 31;    // pooled z:      conv z base = bz*2
  const int bb  = bi >> 10;          // batch

  // ---- stage weights as pre-swizzled 16x32 f16 A-fragments ----
  for (int w = tid; w < NFRAG*16; w += 256) {
    int slot = w & 15;
    int frag = w >> 4;               // (t*2+mt)*32 + lane
    int lane = frag & 31;
    int mt   = (frag >> 5) & 1;
    int t    = frag >> 6;
    int g = lane >> 4;
    int v = slot >> 1, h = slot & 1;
    // A layout (contiguous, matches B): K = g*16 + 2*v + h
    int kk = (g << 4) + (v << 1) + h;   // in-channel
    int o  = (mt << 4) + (lane & 15);                           // out-channel
    lh[WOFF_H + frag*WFRAG_H + slot] = (_Float16)cw[(o*32 + kk)*27 + t];
  }

  // ---- stage input tile + halo (f32 -> f16), layout [z][y][x][ch] ----
  const int gz0 = bz*2 - 1;
  const int gy0 = by*4 - 1;
  const int gx0 = bxh*32 - 1;
  for (int idx = tid; idx < IN_ZE*IN_YE*IN_XE*32; idx += 256) {
    int xx = idx % IN_XE;
    int r  = idx / IN_XE;
    int yy = r % IN_YE;
    int r2 = r / IN_YE;
    int zz = r2 & 3;
    int ci = r2 >> 2;
    int gz = gz0 + zz, gy = gy0 + yy, gx = gx0 + xx;
    float v = 0.f;                       // SAME zero padding
    if ((unsigned)gz < 64u && (unsigned)gy < 64u && (unsigned)gx < 64u)
      v = x[(((bb*32 + ci)*64 + gz)*64 + gy)*64 + gx];
    lh[((zz*IN_YE + yy)*IN_XE + xx)*PT_H + ci] = (_Float16)v;
  }
  __syncthreads();

  // ---- implicit GEMM: M=32 (2 tiles), N=256 conv pts (8 rows x 32 cols), K=864 ----
  const int lane = tid & 31;
  const int wave = tid >> 5;           // wave == conv row: zz_c = wave>>2, yy_c = wave&3
  const int lo = lane & 15, hi = lane >> 4;
  const int zz_c = wave >> 2, yy_c = wave & 3;

  // Per-lane base pointers; all per-tap offsets become compile-time
  // immediates after full unroll (fit in 16-bit DS offsets).
  const uint32_t* wb    = lu + WOFF_U + lane*WFRAG_U;
  const uint32_t* bbase = lu + ((zz_c*IN_YE + yy_c)*IN_XE + lo)*PT_U + hi*8;

  // K split into even/odd taps: 8 independent WMMA accumulation chains.
  v8f acc[2][2][2] = {};               // [xtile j][mtile][t&1]

#pragma unroll
  for (int t = 0; t < 27; ++t) {
    const int dz = t / 9, dy = (t / 3) % 3, dx = t % 3;  // constants after unroll
    const int p  = t & 1;
    union { v16h v; uint32_t u[8]; } a0, a1, b0, b1;
    const int woff = t*2*32*WFRAG_U;                     // uints, constant
    const int boff = ((dz*IN_YE + dy)*IN_XE + dx)*PT_U;  // uints, constant
#pragma unroll
    for (int q = 0; q < 8; ++q) {
      a0.u[q] = wb[woff + q];
      a1.u[q] = wb[woff + 32*WFRAG_U + q];
      b0.u[q] = bbase[boff + q];                         // x-tile 0, cols 0..15
      b1.u[q] = bbase[boff + 16*PT_U + q];               // x-tile 1, cols 16..31
    }
    acc[0][0][p] = __builtin_amdgcn_wmma_f32_16x16x32_f16(false, a0.v, false, b0.v, (short)0, acc[0][0][p], false, false);
    acc[0][1][p] = __builtin_amdgcn_wmma_f32_16x16x32_f16(false, a1.v, false, b0.v, (short)0, acc[0][1][p], false, false);
    acc[1][0][p] = __builtin_amdgcn_wmma_f32_16x16x32_f16(false, a0.v, false, b1.v, (short)0, acc[1][0][p], false, false);
    acc[1][1][p] = __builtin_amdgcn_wmma_f32_16x16x32_f16(false, a1.v, false, b1.v, (short)0, acc[1][1][p], false, false);
  }

  // ---- fused avgpool epilogue: C frags -> LDS pool[o][row][x] -> D ----
  __syncthreads();                     // input region no longer needed; alias it
  float* pool = (float*)smem;          // 32*8*32 f32 = 32 KB
#pragma unroll
  for (int j = 0; j < 2; ++j)
#pragma unroll
    for (int mt = 0; mt < 2; ++mt) {
      v8f c = acc[j][mt][0] + acc[j][mt][1];
#pragma unroll
      for (int r = 0; r < 8; ++r) {
        // C layout: VGPR r, lanes 0-15: M=r, N=lane; lanes 16-31: M=8+r
        int o  = (mt << 4) + (hi << 3) + r;
        int xl = (j << 4) + lo;
        pool[(o*8 + wave)*32 + xl] = c[r];
      }
    }
  __syncthreads();

  for (int q = tid; q < 32*2*16; q += 256) {     // 32 ch x 2 py x 16 px
    int pxl = q & 15;
    int py  = (q >> 4) & 1;
    int o   = q >> 5;
    float s = 0.f;
#pragma unroll
    for (int zz = 0; zz < 2; ++zz)
#pragma unroll
      for (int yy = 0; yy < 2; ++yy)
#pragma unroll
        for (int xx = 0; xx < 2; ++xx)
          s += pool[(o*8 + (zz*4 + py*2 + yy))*32 + (pxl*2 + xx)];
    float dv = s * 0.125f + cb[o];
    Dout[(((bb*32 + o)*32 + bz)*32 + (by*2 + py))*32 + (bxh*16 + pxl)] = dv;
  }
}

// =====================================================================
// Kernel 2: channel sum S[b,z,y,x] = sum_c D  (rank-1 gaussian trick)
// =====================================================================
__global__ __launch_bounds__(256) void chan_sum(const float* __restrict__ D,
                                                float* __restrict__ S)
{
  int idx = blockIdx.x*256 + threadIdx.x;        // 65536
  int b = idx >> 15, s = idx & 32767;
  float acc = 0.f;
#pragma unroll
  for (int c = 0; c < 32; ++c) acc += D[(b*32 + c)*32768 + s];
  S[idx] = acc;
}

// =====================================================================
// Kernel 3: 4 fused 5^3 gaussian convs over S -> normalized clipped grid
// (grids identical across channels, so this IS mean(grids, axis=1))
// =====================================================================
__global__ __launch_bounds__(256) void gauss_grid(const float* __restrict__ S,
    const float* __restrict__ gw, float* __restrict__ G)
{
  __shared__ float gk[125];
  if (threadIdx.x < 125) gk[threadIdx.x] = gw[threadIdx.x];  // gaus_w[0][0][...]
  __syncthreads();
  int idx = blockIdx.x*256 + threadIdx.x;        // 65536
  int b = idx >> 15, s = idx & 32767;
  int z = s >> 10, y = (s >> 5) & 31, xv = s & 31;
  const float inv31 = 1.f/31.f;
  float p = 0.f, f0 = 0.f, f1 = 0.f, f2 = 0.f;
  for (int tz = 0; tz < 5; ++tz) {
    int sz = z + tz - 2; if ((unsigned)sz >= 32u) continue;
    for (int ty = 0; ty < 5; ++ty) {
      int sy = y + ty - 2; if ((unsigned)sy >= 32u) continue;
#pragma unroll
      for (int tx = 0; tx < 5; ++tx) {
        int sx = xv + tx - 2; if ((unsigned)sx >= 32u) continue;
        float g  = gk[(tz*5 + ty)*5 + tx];
        float sv = S[b*32768 + (sz << 10) + (sy << 5) + sx];
        float gs = g * sv;
        p  += gs;
        f0 += gs * (sz * inv31);
        f1 += gs * (sy * inv31);
        f2 += gs * (sx * inv31);
      }
    }
  }
  float inv = 1.f / (p + 1e-6f);
  float g0 = fminf(1.f, fmaxf(-1.f, f0*inv*2.f - 1.f));
  float g1 = fminf(1.f, fmaxf(-1.f, f1*inv*2.f - 1.f));
  float g2 = fminf(1.f, fmaxf(-1.f, f2*inv*2.f - 1.f));
  G[idx*3 + 0] = g0; G[idx*3 + 1] = g1; G[idx*3 + 2] = g2;
}

// =====================================================================
// Kernel 4: trilinear grid sample of x (64^3 vols) at the shared grid
// =====================================================================
__global__ __launch_bounds__(256) void grid_sample3d(const float* __restrict__ x,
    const float* __restrict__ G, float* __restrict__ y)
{
  int idx = blockIdx.x*256 + threadIdx.x;        // 2097152
  int s  = idx & 32767;
  int bc = idx >> 15;                            // b*32 + c
  int b  = bc >> 5;
  const float* gp = G + (size_t)(b*32768 + s)*3;
  float ix = ((gp[0] + 1.f)*64.f - 1.f)*0.5f;    // grid[...,0] -> W (last dim)
  float iy = ((gp[1] + 1.f)*64.f - 1.f)*0.5f;
  float iz = ((gp[2] + 1.f)*64.f - 1.f)*0.5f;
  float x0f = floorf(ix), y0f = floorf(iy), z0f = floorf(iz);
  float wx = ix - x0f, wy = iy - y0f, wz = iz - z0f;
  int x0 = (int)x0f, y0 = (int)y0f, z0 = (int)z0f;
  const float* vol = x + ((size_t)bc << 18);
  float acc = 0.f;
#pragma unroll
  for (int dz = 0; dz < 2; ++dz)
#pragma unroll
    for (int dy = 0; dy < 2; ++dy)
#pragma unroll
      for (int dx = 0; dx < 2; ++dx) {
        int zt = z0 + dz, yt = y0 + dy, xt = x0 + dx;
        float w = (dz ? wz : 1.f - wz) * (dy ? wy : 1.f - wy) * (dx ? wx : 1.f - wx);
        if ((unsigned)zt < 64u && (unsigned)yt < 64u && (unsigned)xt < 64u)
          acc += w * vol[(zt << 12) + (yt << 6) + xt];
      }
  y[idx] = acc;
}

// =====================================================================
extern "C" void kernel_launch(void* const* d_in, const int* in_sizes, int n_in,
                              void* d_out, int out_size, void* d_ws, size_t ws_size,
                              hipStream_t stream)
{
  const float* x      = (const float*)d_in[0];
  const float* conv_w = (const float*)d_in[1];
  const float* conv_b = (const float*)d_in[2];
  const float* gaus_w = (const float*)d_in[3];

  float* out  = (float*)d_out;
  float* yOut = out;                 // (2,32,32,32,32)
  float* DOut = out + Y_ELEMS;       // (2,32,32,32,32)
  float* GOut = out + Y_ELEMS + D_ELEMS;  // (2,32,32,32,3) == mean(grids)
  float* S    = (float*)d_ws;        // 65536 f32 channel-sum scratch

  conv_pool_wmma<<<dim3(2048), dim3(256), LDS_BYTES, stream>>>(x, conv_w, conv_b, DOut);
  chan_sum      <<<dim3(256),  dim3(256), 0, stream>>>(DOut, S);
  gauss_grid    <<<dim3(256),  dim3(256), 0, stream>>>(S, gaus_w, GOut);
  grid_sample3d <<<dim3(8192), dim3(256), 0, stream>>>(x, GOut, yOut);
}
